// Attention_2508260901295
// MI455X (gfx1250) — compile-verified, matrix operand order checked by reading, not run on hardware
//
#include <hip/hip_runtime.h>
#include <hip/hip_bf16.h>

// ---------------------------------------------------------------------------
// Fused attention block for MI455X (gfx1250, wave32, WMMA).
//   B=4, N=2048, D=1024, H=8, HD=128
//   Stage 1: qkv = x @ Wkv^T  (f32 in, f16 to ws, Q pre-scaled by 1/sqrt(128))
//   Stage 2: flash attention per (b,h), online softmax, f16 WMMA, f32 acc,
//            K/Q tiles staged via GLOBAL_LOAD_ASYNC_TO_LDS when available
//   Stage 3: out = y @ Wo^T + bo (f32 out)
// ---------------------------------------------------------------------------

typedef __attribute__((ext_vector_type(4)))  _Float16 h4;
typedef __attribute__((ext_vector_type(8)))  _Float16 h8;
typedef __attribute__((ext_vector_type(16))) _Float16 h16;
typedef __attribute__((ext_vector_type(8)))  float    f8;

#define BATCH 4
#define SEQ   2048
#define DMODEL 1024
#define NHEAD 8
#define HDIM  128

// --- gfx1250 async global->LDS copy (ASYNCcnt-tracked), with fallback -------
#if defined(__gfx1250__) && __has_builtin(__builtin_amdgcn_global_load_async_to_lds_b128)
#define HAVE_ASYNC_LDS 1
// Parameter type revealed by compiler diagnostic: pointer to <4 x i32> vector.
typedef __attribute__((__vector_size__(4 * sizeof(int)))) int i4v;
static __device__ __forceinline__ void async_cp16(const void* g, void* l) {
  __builtin_amdgcn_global_load_async_to_lds_b128((i4v*)g, (i4v*)l, 0, 0);
}
#else
#define HAVE_ASYNC_LDS 0
static __device__ __forceinline__ void async_cp16(const void* g, void* l) {
  *(h8*)l = *(const h8*)g;
}
#endif

static __device__ __forceinline__ void async_wait_all() {
#if HAVE_ASYNC_LDS
#if __has_builtin(__builtin_amdgcn_s_wait_asynccnt)
  __builtin_amdgcn_s_wait_asynccnt(0);
#else
  asm volatile("s_wait_asynccnt 0" ::: "memory");
#endif
#endif
}

// v_wmma_f32_16x16x32_f16 wrapper (8-arg probe-confirmed signature)
static __device__ __forceinline__ f8 wmma32(h16 a, h16 b, f8 c) {
  return __builtin_amdgcn_wmma_f32_16x16x32_f16(false, a, false, b,
                                                (short)0, c, false, false);
}

// Load a 16x32 f16 A/B fragment from a row-major LDS tile.
// CDNA5 16-bit A layout: lane L (L%16 = matrix row, L/16 = k-group) holds
// K = kg*8 + {0..7} in elements 0..7 and K = 16 + kg*8 + {0..7} in elements
// 8..15  ->  two contiguous 16-byte LDS loads per lane.
static __device__ __forceinline__ h16 frag_ld(const _Float16* lds, int row,
                                              int kbase, int ld) {
  const int lane = threadIdx.x & 31;
  const _Float16* p = lds + (row + (lane & 15)) * ld + kbase + ((lane >> 4) << 3);
  h8 lo = *(const h8*)(p);
  h8 hi = *(const h8*)(p + 16);
  return __builtin_shufflevector(lo, hi, 0, 1, 2, 3, 4, 5, 6, 7, 8, 9, 10, 11,
                                 12, 13, 14, 15);
}

// ---------------------------------------------------------------------------
// Kernel 1: QKV projection. grid = (3*D/128, B*N/128), block = 256 (8 waves).
// Each wave computes a 64x32 register tile (8 wmma accumulators).
// ---------------------------------------------------------------------------
__global__ __launch_bounds__(256) void qkv_proj_kernel(
    const float* __restrict__ x, const float* __restrict__ Wkv,
    _Float16* __restrict__ qws, _Float16* __restrict__ kws,
    _Float16* __restrict__ vws) {
  extern __shared__ _Float16 smem[];
  const int LD = 40;  // 32 + 8 pad (halfs)
  _Float16* Xt = smem;            // [128][40]
  _Float16* Wt = smem + 128 * LD; // [128][40]

  const int tid = threadIdx.x;
  const int wave = tid >> 5;
  const int lane = tid & 31;
  const int nloc = lane & 15;
  const int mhi = (lane >> 4) << 3;  // 0 or 8
  const int m0 = blockIdx.y * 128;
  const int n0 = blockIdx.x * 128;
  const int wm = wave >> 2;  // 0..1 -> 64-row half
  const int wn = wave & 3;   // 0..3 -> 32-col quarter

  f8 acc[4][2] = {};

  for (int k0 = 0; k0 < DMODEL; k0 += 32) {
    __syncthreads();
    // 128x32 tile = 1024 float4 chunks; 4 per thread, converted to f16
#pragma unroll
    for (int i = 0; i < 4; ++i) {
      int chunk = tid + i * 256;
      int r = chunk >> 3, c4 = (chunk & 7) * 4;
      float4 fx = *(const float4*)(x + (size_t)(m0 + r) * DMODEL + k0 + c4);
      float4 fw = *(const float4*)(Wkv + (size_t)(n0 + r) * DMODEL + k0 + c4);
      h4 hx = {(_Float16)fx.x, (_Float16)fx.y, (_Float16)fx.z, (_Float16)fx.w};
      h4 hw = {(_Float16)fw.x, (_Float16)fw.y, (_Float16)fw.z, (_Float16)fw.w};
      *(h4*)(Xt + r * LD + c4) = hx;
      *(h4*)(Wt + r * LD + c4) = hw;
    }
    __syncthreads();

    h16 a[4], b[2];
#pragma unroll
    for (int i = 0; i < 4; ++i) a[i] = frag_ld(Xt, wm * 64 + i * 16, 0, LD);
#pragma unroll
    for (int j = 0; j < 2; ++j) b[j] = frag_ld(Wt, wn * 32 + j * 16, 0, LD);
#pragma unroll
    for (int i = 0; i < 4; ++i)
#pragma unroll
      for (int j = 0; j < 2; ++j) acc[i][j] = wmma32(a[i], b[j], acc[i][j]);
  }

  // Epilogue: destination (q/k/v, head) is uniform per 16-wide tile.
#pragma unroll
  for (int j = 0; j < 2; ++j) {
    int gnb = n0 + wn * 32 + j * 16;        // tile base col (multiple of 16)
    int which = gnb >> 10;                  // 0=q 1=k 2=v (uniform)
    int e = gnb & (DMODEL - 1);
    int hh = e >> 7;                        // head (uniform)
    int hdb = (e & (HDIM - 1)) + nloc;      // feature within head
    _Float16* dst = (which == 0) ? qws : (which == 1) ? kws : vws;
    float scale = (which == 0) ? 0.08838834764831845f : 1.0f;
#pragma unroll
    for (int i = 0; i < 4; ++i)
#pragma unroll
      for (int r = 0; r < 8; ++r) {
        int gm = m0 + wm * 64 + i * 16 + mhi + r;  // row in [0, B*N)
        int bb = gm >> 11, nn = gm & (SEQ - 1);
        dst[(((size_t)bb * NHEAD + hh) * SEQ + nn) * HDIM + hdb] =
            (_Float16)(acc[i][j][r] * scale);
      }
  }
}

// ---------------------------------------------------------------------------
// Kernel 2: flash attention. grid = (B*H, N/128), block = 256 (8 waves).
// Wave w owns query rows [w*16, w*16+16) of the 128-query tile.
// ---------------------------------------------------------------------------
__global__ __launch_bounds__(256) void flash_attn_kernel(
    const _Float16* __restrict__ qws, const _Float16* __restrict__ kws,
    const _Float16* __restrict__ vws, const float* __restrict__ mask,
    _Float16* __restrict__ yws) {
  extern __shared__ _Float16 smem[];
  const int LD = 136;  // 128 + 8 pad
  _Float16* Qt = smem;                 // [128][136] row-major (q x d)
  _Float16* Kt = smem + 128 * LD;      // [128][136] row-major (kv x d)
  _Float16* Vt = smem + 2 * 128 * LD;  // [128][136] TRANSPOSED (d x kv)
  _Float16* Pt = smem + 3 * 128 * LD;  // [128][136] probs (q x kv)

  const int tid = threadIdx.x;
  const int w = tid >> 5;
  const int lane = tid & 31;
  const int nloc = lane & 15;
  const int mhi = (lane >> 4) << 3;
  const int bh = blockIdx.x;       // b*H + h
  const int q0 = blockIdx.y * 128;
  const size_t hbase = (size_t)bh * SEQ * HDIM;

  // Q tile: 2048 x 16B chunks, 8 per thread (async when available)
#pragma unroll
  for (int i = 0; i < 8; ++i) {
    int chunk = tid + i * 256;
    int r = chunk >> 4, c8 = (chunk & 15) * 8;
    async_cp16(qws + hbase + (size_t)(q0 + r) * HDIM + c8, Qt + r * LD + c8);
  }

  f8 o[8] = {};
  float mrow[8], lrow[8];
#pragma unroll
  for (int r = 0; r < 8; ++r) { mrow[r] = -1.0e30f; lrow[r] = 0.0f; }

  for (int kv0 = 0; kv0 < SEQ; kv0 += 128) {
    async_wait_all();   // our async ops (Q and previous K) retired
    __syncthreads();    // previous tile fully consumed by all waves
    // K tile: async row-major copy; V tile: load h8 + transpose scatter
#pragma unroll
    for (int i = 0; i < 8; ++i) {
      int chunk = tid + i * 256;
      int r = chunk >> 4, c8 = (chunk & 15) * 8;
      async_cp16(kws + hbase + (size_t)(kv0 + r) * HDIM + c8, Kt + r * LD + c8);
      h8 vv = *(const h8*)(vws + hbase + (size_t)(kv0 + r) * HDIM + c8);
#pragma unroll
      for (int e2 = 0; e2 < 8; ++e2) Vt[(c8 + e2) * LD + r] = vv[e2];
    }
    __builtin_prefetch(&mask[(size_t)(q0 + w * 16) * SEQ + kv0], 0, 1);
    async_wait_all();
    __syncthreads();

    // S = Q @ K^T   (B = K^T  ->  same row-major fragment pattern as A)
    f8 s[8] = {};
#pragma unroll
    for (int kk = 0; kk < HDIM; kk += 32) {
      h16 a = frag_ld(Qt, w * 16, kk, LD);
#pragma unroll
      for (int t = 0; t < 8; ++t) {
        h16 b = frag_ld(Kt, t * 16, kk, LD);
        s[t] = wmma32(a, b, s[t]);
      }
    }

    // additive mask
    const float* mp = mask + (size_t)(q0 + w * 16 + mhi) * SEQ + kv0 + nloc;
#pragma unroll
    for (int t = 0; t < 8; ++t)
#pragma unroll
      for (int r = 0; r < 8; ++r) s[t][r] += mp[(size_t)r * SEQ + t * 16];

    // online softmax: each row lives across 16 lanes of one half-wave
#pragma unroll
    for (int r = 0; r < 8; ++r) {
      float tm = s[0][r];
#pragma unroll
      for (int t = 1; t < 8; ++t) tm = fmaxf(tm, s[t][r]);
      tm = fmaxf(tm, __shfl_xor(tm, 1, 32));
      tm = fmaxf(tm, __shfl_xor(tm, 2, 32));
      tm = fmaxf(tm, __shfl_xor(tm, 4, 32));
      tm = fmaxf(tm, __shfl_xor(tm, 8, 32));
      float nm = fmaxf(mrow[r], tm);
      float corr = __expf(mrow[r] - nm);
      mrow[r] = nm;
      float ps = 0.0f;
#pragma unroll
      for (int t = 0; t < 8; ++t) {
        float p = __expf(s[t][r] - nm);
        ps += p;
        Pt[(w * 16 + mhi + r) * LD + t * 16 + nloc] = (_Float16)p;
      }
      ps += __shfl_xor(ps, 1, 32);
      ps += __shfl_xor(ps, 2, 32);
      ps += __shfl_xor(ps, 4, 32);
      ps += __shfl_xor(ps, 8, 32);
      lrow[r] = lrow[r] * corr + ps;
#pragma unroll
      for (int t = 0; t < 8; ++t) o[t][r] *= corr;
    }
    // Pt rows are wave-private; same-wave LDS ops are in-order -> no barrier.

    // O += P @ V   (B from transposed Vt -> row-major fragment pattern)
#pragma unroll
    for (int kk = 0; kk < 128; kk += 32) {
      h16 a = frag_ld(Pt, w * 16, kk, LD);
#pragma unroll
      for (int t = 0; t < 8; ++t) {
        h16 b = frag_ld(Vt, t * 16, kk, LD);
        o[t] = wmma32(a, b, o[t]);
      }
    }
  }

  // normalize and write y [B][N][D] (f16)
  const int b = bh >> 3, h = bh & 7;
#pragma unroll
  for (int t = 0; t < 8; ++t)
#pragma unroll
    for (int r = 0; r < 8; ++r) {
      float val = o[t][r] / lrow[r];
      int row = q0 + w * 16 + mhi + r;
      int col = h * HDIM + t * 16 + nloc;
      yws[((size_t)b * SEQ + row) * DMODEL + col] = (_Float16)val;
    }
}

// ---------------------------------------------------------------------------
// Kernel 3: output projection out = y @ Wo^T + bo. grid = (D/128, B*N/128).
// ---------------------------------------------------------------------------
__global__ __launch_bounds__(256) void out_proj_kernel(
    const _Float16* __restrict__ yws, const float* __restrict__ Wo,
    const float* __restrict__ bo, float* __restrict__ out) {
  extern __shared__ _Float16 smem[];
  const int LD = 40;
  _Float16* Yt = smem;
  _Float16* Wt = smem + 128 * LD;

  const int tid = threadIdx.x;
  const int wave = tid >> 5;
  const int lane = tid & 31;
  const int nloc = lane & 15;
  const int mhi = (lane >> 4) << 3;
  const int m0 = blockIdx.y * 128;
  const int n0 = blockIdx.x * 128;
  const int wm = wave >> 2;
  const int wn = wave & 3;

  f8 acc[4][2] = {};

  for (int k0 = 0; k0 < DMODEL; k0 += 32) {
    __syncthreads();
    // A tile is f16 already: 512 h8 chunks (2/thread); W tile: float4 convert
#pragma unroll
    for (int i = 0; i < 2; ++i) {
      int chunk = tid + i * 256;
      int r = chunk >> 2, c8 = (chunk & 3) * 8;
      *(h8*)(Yt + r * LD + c8) =
          *(const h8*)(yws + (size_t)(m0 + r) * DMODEL + k0 + c8);
    }
#pragma unroll
    for (int i = 0; i < 4; ++i) {
      int chunk = tid + i * 256;
      int r = chunk >> 3, c4 = (chunk & 7) * 4;
      float4 fw = *(const float4*)(Wo + (size_t)(n0 + r) * DMODEL + k0 + c4);
      h4 hw = {(_Float16)fw.x, (_Float16)fw.y, (_Float16)fw.z, (_Float16)fw.w};
      *(h4*)(Wt + r * LD + c4) = hw;
    }
    __syncthreads();

    h16 a[4], b[2];
#pragma unroll
    for (int i = 0; i < 4; ++i) a[i] = frag_ld(Yt, wm * 64 + i * 16, 0, LD);
#pragma unroll
    for (int j = 0; j < 2; ++j) b[j] = frag_ld(Wt, wn * 32 + j * 16, 0, LD);
#pragma unroll
    for (int i = 0; i < 4; ++i)
#pragma unroll
      for (int j = 0; j < 2; ++j) acc[i][j] = wmma32(a[i], b[j], acc[i][j]);
  }

#pragma unroll
  for (int i = 0; i < 4; ++i)
#pragma unroll
    for (int j = 0; j < 2; ++j)
#pragma unroll
      for (int r = 0; r < 8; ++r) {
        int gm = m0 + wm * 64 + i * 16 + mhi + r;
        int gn = n0 + wn * 32 + j * 16 + nloc;
        out[(size_t)gm * DMODEL + gn] = acc[i][j][r] + bo[gn];
      }
}

// ---------------------------------------------------------------------------
// Host launch. Inputs: x, mask, Wkv, Wo, bo (all f32). Output: f32 [B,N,D].
// Workspace layout (f16): Q | K | V | Y, each B*N*D elements (64 MB total).
// ---------------------------------------------------------------------------
extern "C" void kernel_launch(void* const* d_in, const int* in_sizes, int n_in,
                              void* d_out, int out_size, void* d_ws,
                              size_t ws_size, hipStream_t stream) {
  (void)in_sizes; (void)n_in; (void)out_size; (void)ws_size;
  const float* x    = (const float*)d_in[0];
  const float* mask = (const float*)d_in[1];
  const float* Wkv  = (const float*)d_in[2];
  const float* Wo   = (const float*)d_in[3];
  const float* bo   = (const float*)d_in[4];
  float* out = (float*)d_out;

  const size_t SZ = (size_t)BATCH * SEQ * DMODEL;  // 8.4M elements
  _Float16* base = (_Float16*)d_ws;
  _Float16* qws = base;
  _Float16* kws = base + SZ;
  _Float16* vws = base + 2 * SZ;
  _Float16* yws = base + 3 * SZ;

  const size_t lds_gemm = 2 * 128 * 40 * sizeof(_Float16);   // 20.5 KB
  const size_t lds_attn = 4 * 128 * 136 * sizeof(_Float16);  // 136 KB (of 320)

  qkv_proj_kernel<<<dim3(3 * DMODEL / 128, BATCH * SEQ / 128), 256, lds_gemm,
                    stream>>>(x, Wkv, qws, kws, vws);
  flash_attn_kernel<<<dim3(BATCH * NHEAD, SEQ / 128), 256, lds_attn, stream>>>(
      qws, kws, vws, mask, yws);
  out_proj_kernel<<<dim3(DMODEL / 128, BATCH * SEQ / 128), 256, lds_gemm,
                    stream>>>(yws, Wo, bo, out);
}
